// xLSTMRegressor_12610023981079
// MI455X (gfx1250) — matrix operand order repaired and checked by reading, not compiled, not run on hardware
//
#include <hip/hip_runtime.h>
#include <hip/hip_bf16.h>
#include <math.h>

// ---------------------------------------------------------------------------
// xLSTM regressor forward for MI455X (gfx1250, wave32, WMMA).
// Heavy GEMMs -> LDS-staged, double-buffered v_wmma_f32_16x16x32_bf16.
// sLSTM recurrence -> 256KB fp32 recurrent matrix cached in the 320KB WGP LDS.
// ---------------------------------------------------------------------------

#define BB      128
#define SS      64
#define T_TOK   (BB * SS)      // 8192 tokens
#define IN_F    3000
#define IN_FP   3008           // padded to multiple of 32 for WMMA K loop
#define EE      512
#define INNER_  1024
#define NHM     4
#define DHM     256
#define NQK_    256
#define NHS     4
#define DHS_    128
#define FF_     704

typedef __bf16 bf16;
typedef __attribute__((ext_vector_type(8)))  __bf16 v8bf;
typedef __attribute__((ext_vector_type(16))) __bf16 v16bf;
typedef __attribute__((ext_vector_type(8)))  float  v8f;

union BfFrag { v16bf v; v8bf h[2]; };

// ---------------- device math helpers ----------------
__device__ __forceinline__ float silu_f(float x)   { return x / (1.f + expf(-x)); }
__device__ __forceinline__ float sigm_f(float x)   { return 1.f / (1.f + expf(-x)); }
__device__ __forceinline__ float logsig_f(float x) { return fminf(x, 0.f) - log1pf(expf(-fabsf(x))); }
__device__ __forceinline__ float gelu_f(float x)   { return 0.5f * x * (1.f + erff(x * 0.70710678118654752f)); }

// ---------------- WMMA fragment loads (ISA 7.12.2 bf16 layouts) ----------------
// A 16x32 (MxK): lanes 0-15 = rows M, hold K 0..7 (v0-3) and 16..23 (v4-7);
// lanes 16-31 same rows, K 8..15 and 24..31.  ld = row stride in elements.
__device__ __forceinline__ v16bf frag_a(const bf16* __restrict__ base, int ld, int lane) {
    int r = lane & 15, half = lane >> 4;
    const bf16* row = base + (size_t)r * ld;
    BfFrag f;
    f.h[0] = *(const v8bf*)(row + half * 8);
    f.h[1] = *(const v8bf*)(row + 16 + half * 8);
    return f.v;
}
// B 32x16 (KxN) from weight rows [N,K] (B[k][n]=W[n][k]):
// lane&15 = column n; lanes 0-15 hold K 0..7 and 16..23, lanes 16-31 hold K 8..15 and 24..31.
__device__ __forceinline__ v16bf frag_b(const bf16* __restrict__ base, int ld, int lane) {
    int n = lane & 15, kh = lane >> 4;
    const bf16* row = base + (size_t)n * ld;
    BfFrag f;
    f.h[0] = *(const v8bf*)(row + kh * 8);
    f.h[1] = *(const v8bf*)(row + 16 + kh * 8);
    return f.v;
}

// ---------------- GEMM: C[M,N] = A[M,K](bf16) * W[N,K](bf16)^T (+bias)(+residual)
// Block = 256 threads = 8 waves; block tile 128x128, K-step 32, double-buffered
// LDS slabs (A,B: 128 rows x 32 bf16, rows padded to 40 elems for bank spread).
// Wave (wm = w&3, wn = w>>2) owns a 32x64 strip = 2x4 WMMA tiles.
// Requires M%128==0, N%128==0, K%32==0.
#define LDS_P 40
__global__ __launch_bounds__(256) void gemm_bf16_wmma_kernel(
    const bf16* __restrict__ A, int lda,
    const bf16* __restrict__ W, int ldb,
    float* __restrict__ C, int ldc,
    const float* __restrict__ bias,    // [N] or nullptr
    const float* __restrict__ resid,   // same layout as C, or nullptr
    int M, int N, int K)
{
    __shared__ bf16 sA[2][128 * LDS_P];
    __shared__ bf16 sB[2][128 * LDS_P];

    const int tid  = threadIdx.x;
    const int lane = tid & 31;
    const int wave = tid >> 5;
    const int wm   = wave & 3;      // 4 waves over M (32 rows each)
    const int wn   = wave >> 2;     // 2 waves over N (64 cols each)
    const int mBlk = blockIdx.y * 128;
    const int nBlk = blockIdx.x * 128;

    // staging tasks: 512 16B chunks per slab -> 2 per thread
    const int row0 = tid >> 1;            // unused helper kept simple below
    (void)row0;

    v8f acc[2][4];
    #pragma unroll
    for (int mi = 0; mi < 2; ++mi)
        #pragma unroll
        for (int nj = 0; nj < 4; ++nj)
            #pragma unroll
            for (int e = 0; e < 8; ++e) acc[mi][nj][e] = 0.f;

    // ---- stage K-slab 0 into buffer 0
    #pragma unroll
    for (int p = 0; p < 2; ++p) {
        int task = tid + p * 256;
        int row = task >> 2, qq = task & 3;
        *(v8bf*)&sA[0][row * LDS_P + qq * 8] =
            *(const v8bf*)(A + (size_t)(mBlk + row) * lda + qq * 8);
        *(v8bf*)&sB[0][row * LDS_P + qq * 8] =
            *(const v8bf*)(W + (size_t)(nBlk + row) * ldb + qq * 8);
    }
    __syncthreads();

    int buf = 0;
    for (int k0 = 0; k0 < K; k0 += 32, buf ^= 1) {
        const bool more = (k0 + 32) < K;
        v8bf ra[2], rb[2];
        if (more) {
            // fetch next slab into registers (overlaps with WMMA below)
            #pragma unroll
            for (int p = 0; p < 2; ++p) {
                int task = tid + p * 256;
                int row = task >> 2, qq = task & 3;
                ra[p] = *(const v8bf*)(A + (size_t)(mBlk + row) * lda + k0 + 32 + qq * 8);
                rb[p] = *(const v8bf*)(W + (size_t)(nBlk + row) * ldb + k0 + 32 + qq * 8);
            }
            if (k0 + 64 < K) {  // hint slab-after-next into cache
                int row = tid >> 2, qq = tid & 3;
                __builtin_prefetch(A + (size_t)(mBlk + row) * lda + k0 + 64 + qq * 8, 0, 3);
                __builtin_prefetch(W + (size_t)(nBlk + row) * ldb + k0 + 64 + qq * 8, 0, 3);
            }
        }

        // ---- compute from LDS buffer `buf`
        v16bf af[2];
        #pragma unroll
        for (int mi = 0; mi < 2; ++mi)
            af[mi] = frag_a(&sA[buf][(wm * 32 + mi * 16) * LDS_P], LDS_P, lane);
        #pragma unroll
        for (int nj = 0; nj < 4; ++nj) {
            v16bf bf_ = frag_b(&sB[buf][(wn * 64 + nj * 16) * LDS_P], LDS_P, lane);
            #pragma unroll
            for (int mi = 0; mi < 2; ++mi)
                acc[mi][nj] = __builtin_amdgcn_wmma_f32_16x16x32_bf16(
                    false, af[mi], false, bf_, (short)0, acc[mi][nj], false, false);
        }

        if (more) {
            // ---- store next slab into the other buffer, then one barrier
            #pragma unroll
            for (int p = 0; p < 2; ++p) {
                int task = tid + p * 256;
                int row = task >> 2, qq = task & 3;
                *(v8bf*)&sA[buf ^ 1][row * LDS_P + qq * 8] = ra[p];
                *(v8bf*)&sB[buf ^ 1][row * LDS_P + qq * 8] = rb[p];
            }
            __syncthreads();
        }
    }

    // ---- epilogue: C/D layout (VGPR e -> M = e, +8 for lanes 16-31)
    const int half = lane >> 4;
    #pragma unroll
    for (int mi = 0; mi < 2; ++mi) {
        #pragma unroll
        for (int nj = 0; nj < 4; ++nj) {
            int ncol = nBlk + wn * 64 + nj * 16 + (lane & 15);
            int mbase = mBlk + wm * 32 + mi * 16 + 8 * half;
            #pragma unroll
            for (int e = 0; e < 8; ++e) {
                size_t idx = (size_t)(mbase + e) * ldc + ncol;
                float v = acc[mi][nj][e];
                if (bias)  v += bias[ncol];
                if (resid) v += resid[idx];
                C[idx] = v;
            }
        }
    }
}

// ---------------- fp32 -> bf16 pack with K padding ----------------
__global__ void pack_bf16_kernel(bf16* __restrict__ dst, const float* __restrict__ src,
                                 int rows, int K, int Kp)
{
    long long i = (long long)blockIdx.x * 256 + threadIdx.x;
    long long total = (long long)rows * Kp;
    if (i >= total) return;
    int c = (int)(i % Kp);
    long long r = i / Kp;
    float v = (c < K) ? src[r * K + c] : 0.f;
    dst[i] = (bf16)v;
}

// ---------------- LayerNorm (plain or per-head weight) ----------------
__global__ __launch_bounds__(256) void ln_kernel(
    float* __restrict__ y, const float* __restrict__ x,
    const float* __restrict__ w, int D, int nh /*0 = plain*/)
{
    __shared__ float red[256];
    long long row = blockIdx.x;
    const float* xr = x + row * D;
    float s = 0.f;
    for (int c = threadIdx.x; c < D; c += 256) s += xr[c];
    red[threadIdx.x] = s; __syncthreads();
    for (int st = 128; st > 0; st >>= 1) { if (threadIdx.x < st) red[threadIdx.x] += red[threadIdx.x + st]; __syncthreads(); }
    float mu = red[0] / D; __syncthreads();
    float s2 = 0.f;
    for (int c = threadIdx.x; c < D; c += 256) { float d0 = xr[c] - mu; s2 += d0 * d0; }
    red[threadIdx.x] = s2; __syncthreads();
    for (int st = 128; st > 0; st >>= 1) { if (threadIdx.x < st) red[threadIdx.x] += red[threadIdx.x + st]; __syncthreads(); }
    float inv = rsqrtf(red[0] / D + 1e-5f);
    const float* wp = nh ? (w + (size_t)(row % nh) * D) : w;
    for (int c = threadIdx.x; c < D; c += 256)
        y[row * D + c] = (xr[c] - mu) * inv * wp[c];
}

// ---------------- causal depthwise conv (K=4) + optional SiLU ----------------
__global__ void conv_silu_kernel(float* __restrict__ y, int ldy,
                                 const float* __restrict__ x, int ldx,
                                 const float* __restrict__ w, const float* __restrict__ bias,
                                 int Cn, int do_silu)
{
    long long idx = (long long)blockIdx.x * 256 + threadIdx.x;
    long long total = (long long)T_TOK * Cn;
    if (idx >= total) return;
    int c = (int)(idx % Cn);
    long long t = idx / Cn;
    int s = (int)(t % SS);
    long long b = t / SS;
    float acc = bias[c];
    #pragma unroll
    for (int j = 0; j < 4; ++j) {
        int sj = s - 3 + j;
        if (sj >= 0) acc += x[(b * SS + sj) * ldx + c] * w[c * 4 + j];
    }
    if (do_silu) acc = silu_f(acc);
    y[t * ldy + c] = acc;
}

// ---------------- 4x4 block-diagonal head projection (q/k/v) ----------------
__global__ void headwise44_kernel(float* __restrict__ y, int ldy,
                                  const float* __restrict__ x, int ldx,
                                  const float* __restrict__ w)
{
    long long idx = (long long)blockIdx.x * 256 + threadIdx.x;
    if (idx >= (long long)T_TOK * NQK_) return;
    int blk = (int)(idx % NQK_);
    long long t = idx / NQK_;
    const float* xr = x + t * ldx + blk * 4;
    const float* wb = w + blk * 16;
    float x0 = xr[0], x1 = xr[1], x2 = xr[2], x3 = xr[3];
    float* yr = y + t * ldy + blk * 4;
    #pragma unroll
    for (int o = 0; o < 4; ++o)
        yr[o] = wb[o*4+0]*x0 + wb[o*4+1]*x1 + wb[o*4+2]*x2 + wb[o*4+3]*x3;
}

// ---------------- ig/fg gate projections: [T] blocks, 8 dots of 3072 each ----
__global__ __launch_bounds__(256) void igfg_kernel(
    float* __restrict__ ig, float* __restrict__ fg,
    const float* __restrict__ q, const float* __restrict__ k, const float* __restrict__ v,
    const float* __restrict__ igw, const float* __restrict__ igb,
    const float* __restrict__ fgw, const float* __restrict__ fgb)
{
    int t = blockIdx.x;
    int dot  = threadIdx.x >> 5;   // 0..7 : (gate, head)
    int lane = threadIdx.x & 31;
    int gsel = dot >> 2, hh = dot & 3;
    const float* w = (gsel ? fgw : igw) + (size_t)hh * 3 * INNER_;
    float acc = 0.f;
    for (int c = lane; c < 3 * INNER_; c += 32) {
        float xv;
        if (c < INNER_)            xv = q[(size_t)t * INNER_ + c];
        else if (c < 2 * INNER_)   xv = k[(size_t)t * INNER_ + (c - INNER_)];
        else                       xv = v[(size_t)t * INNER_ + (c - 2 * INNER_)];
        acc += xv * w[c];
    }
    for (int o = 16; o > 0; o >>= 1) acc += __shfl_down(acc, o, 32);
    if (lane == 0) {
        int b = t / SS, s = t % SS;
        size_t idx = ((size_t)b * NHM + hh) * SS + s;
        if (gsel) fg[idx] = acc + fgb[hh];
        else      ig[idx] = acc + igb[hh];
    }
}

// ---------------- cumulative log-sigmoid of forget gates ----------------
__global__ void lfc_kernel(float* __restrict__ lfc, const float* __restrict__ fg)
{
    int bh = blockIdx.x * blockDim.x + threadIdx.x;
    if (bh >= BB * NHM) return;
    const float* f = fg + (size_t)bh * SS;
    float* o = lfc + (size_t)bh * (SS + 1);
    float run = 0.f; o[0] = 0.f;
    for (int s = 0; s < SS; ++s) { run += logsig_f(f[s]); o[s + 1] = run; }
}

// ---------------- mLSTM parallel core: one block per (b,h), K/V in LDS --------
__global__ __launch_bounds__(256) void mlstm_attn_kernel(
    float* __restrict__ hout,
    const float* __restrict__ q, const float* __restrict__ k, const float* __restrict__ v,
    const float* __restrict__ ig, const float* __restrict__ lfc)
{
    int bh = blockIdx.x, b = bh / NHM, h = bh % NHM;
    extern __shared__ float lds[];
    float* ksm   = lds;                 // SS*DHM
    float* vsm   = ksm + SS * DHM;      // SS*DHM
    float* qrow  = vsm + SS * DHM;      // DHM
    float* sc    = qrow + DHM;          // SS
    float* wj    = sc + SS;             // SS
    float* red   = wj + SS;             // 256
    float* ig_s  = red + 256;           // SS
    float* lfc_s = ig_s + SS;           // SS+1

    for (int i = threadIdx.x; i < SS * DHM; i += 256) {
        int s = i / DHM, d = i % DHM;
        size_t src = ((size_t)(b * SS + s)) * INNER_ + h * DHM + d;
        ksm[i] = k[src]; vsm[i] = v[src];
    }
    if (threadIdx.x < SS)     ig_s[threadIdx.x]  = ig[(size_t)bh * SS + threadIdx.x];
    if (threadIdx.x < SS + 1) lfc_s[threadIdx.x] = lfc[(size_t)bh * (SS + 1) + threadIdx.x];
    __syncthreads();

    const float rsDH = rsqrtf((float)DHM);
    for (int i = 0; i < SS; ++i) {
        for (int d = threadIdx.x; d < DHM; d += 256)
            qrow[d] = q[((size_t)(b * SS + i)) * INNER_ + h * DHM + d];
        __syncthreads();
        // scores: 4 partial threads per column j
        int j = threadIdx.x & 63, part = threadIdx.x >> 6;
        float p = 0.f;
        for (int d = part * 64; d < part * 64 + 64; ++d) p += qrow[d] * ksm[j * DHM + d];
        red[threadIdx.x] = p; __syncthreads();
        if (part == 0) sc[j] = red[j] + red[j + 64] + red[j + 128] + red[j + 192];
        __syncthreads();
        if (threadIdx.x < SS) {
            int jj = threadIdx.x;
            wj[jj] = (jj <= i) ? (lfc_s[i + 1] - lfc_s[jj + 1] + ig_s[jj]) : -1e30f;
        }
        __syncthreads();
        if (threadIdx.x == 0) {
            float mx = -1e30f;
            for (int jj = 0; jj <= i; ++jj) mx = fmaxf(mx, wj[jj]);
            red[0] = mx;
        }
        __syncthreads();
        float maxD = red[0];
        if (threadIdx.x < SS) {
            int jj = threadIdx.x;
            wj[jj] = (jj <= i) ? expf(wj[jj] - maxD) * sc[jj] * rsDH : 0.f;
        }
        __syncthreads();
        if (threadIdx.x == 0) {
            float sm = 0.f;
            for (int jj = 0; jj <= i; ++jj) sm += wj[jj];
            float norm = fmaxf(fabsf(sm), expf(-maxD)) + 1e-6f;
            red[0] = 1.f / norm;
        }
        __syncthreads();
        float invn = red[0];
        for (int d = threadIdx.x; d < DHM; d += 256) {
            float o = 0.f;
            for (int jj = 0; jj <= i; ++jj) o += wj[jj] * vsm[jj * DHM + d];
            hout[((size_t)(b * SS + i)) * INNER_ + h * DHM + d] = o * invn;
        }
        __syncthreads();
    }
}

// ---------------- mLSTM output gating ----------------
__global__ void ew_h2_kernel(float* __restrict__ h2, const float* __restrict__ hln,
                             const float* __restrict__ xc, const float* __restrict__ xi,
                             const float* __restrict__ skip)
{
    long long i = (long long)blockIdx.x * 256 + threadIdx.x;
    if (i >= (long long)T_TOK * INNER_) return;
    int c = (int)(i % INNER_);
    long long t = i / INNER_;
    float z = xi[t * (2 * INNER_) + INNER_ + c];
    h2[i] = (hln[i] + skip[c] * xc[i]) * silu_f(z);
}

__global__ void ew_add_kernel(float* __restrict__ a, const float* __restrict__ b, long long n)
{
    long long i = (long long)blockIdx.x * 256 + threadIdx.x;
    if (i < n) a[i] += b[i];
}

__global__ void ew_geglu_kernel(float* __restrict__ act, const float* __restrict__ u)
{
    long long i = (long long)blockIdx.x * 256 + threadIdx.x;
    if (i >= (long long)T_TOK * FF_) return;
    int c = (int)(i % FF_);
    long long t = i / FF_;
    float a = u[t * (2 * FF_) + c];
    float g = u[t * (2 * FF_) + FF_ + c];
    act[i] = gelu_f(a) * g;
}

// ---------------- sLSTM recurrent scan: recurrent matrix in 256KB LDS ---------
__global__ __launch_bounds__(256) void slstm_scan_kernel(
    float* __restrict__ ys,
    const float* __restrict__ gi, const float* __restrict__ gf,
    const float* __restrict__ gz, const float* __restrict__ go,
    const float* __restrict__ rk, const float* __restrict__ rb)
{
    int bh = blockIdx.x, b = bh >> 2, h = bh & 3;
    extern __shared__ float lds[];
    float* rks = lds;                        // DHS_*4*DHS_ = 65536 floats (256KB, CDNA5 LDS)
    float* yv  = rks + DHS_ * 4 * DHS_;      // DHS_
    float* ry  = yv + DHS_;                  // 4*DHS_
    for (int i = threadIdx.x; i < DHS_ * 4 * DHS_; i += 256)
        rks[i] = rk[(size_t)h * DHS_ * 4 * DHS_ + i];
    if (threadIdx.x < DHS_) yv[threadIdx.x] = 0.f;
    float c_st = 0.f, n_st = 0.f, m_st = 0.f;
    __syncthreads();
    const float* rbh = rb + (size_t)h * 4 * DHS_;
    for (int s = 0; s < SS; ++s) {
        for (int kk = threadIdx.x; kk < 4 * DHS_; kk += 256) {
            float acc = 0.f;
            for (int d = 0; d < DHS_; ++d) acc += yv[d] * rks[d * 4 * DHS_ + kk];
            ry[kk] = acc;
        }
        __syncthreads();
        if (threadIdx.x < DHS_) {
            int d = threadIdx.x;
            size_t base = ((size_t)(b * SS + s)) * EE + h * DHS_ + d;
            float ir = gi[base] + ry[0 * DHS_ + d] + rbh[0 * DHS_ + d];
            float fr = gf[base] + ry[1 * DHS_ + d] + rbh[1 * DHS_ + d];
            float zr = gz[base] + ry[2 * DHS_ + d] + rbh[2 * DHS_ + d];
            float og = go[base] + ry[3 * DHS_ + d] + rbh[3 * DHS_ + d];
            float lpm  = m_st + logsig_f(fr);
            float mnew = (s == 0) ? ir : fmaxf(ir, lpm);
            float ie = expf(ir - mnew);
            float fe = expf(lpm - mnew);
            float cn = fe * c_st + ie * tanhf(zr);
            float nn = fe * n_st + ie;
            float yn = sigm_f(og) * cn / nn;
            c_st = cn; n_st = nn; m_st = mnew;
            yv[d] = yn;
            ys[base] = yn;
        }
        __syncthreads();
    }
}

// ---------------- final projection + softplus ----------------
__global__ __launch_bounds__(256) void fc_softplus_kernel(
    float* __restrict__ out, const float* __restrict__ hfin,
    const float* __restrict__ fcw, const float* __restrict__ fcb)
{
    __shared__ float red[256];
    int b = blockIdx.x;
    const float* row = hfin + ((size_t)(b * SS + SS - 1)) * EE;
    float acc = 0.f;
    for (int c = threadIdx.x; c < EE; c += 256) acc += row[c] * fcw[c];
    red[threadIdx.x] = acc; __syncthreads();
    for (int st = 128; st > 0; st >>= 1) { if (threadIdx.x < st) red[threadIdx.x] += red[threadIdx.x + st]; __syncthreads(); }
    if (threadIdx.x == 0) {
        float z = red[0] + fcb[0];
        out[b] = (z > 20.f) ? z : log1pf(expf(z));
    }
}

// ===========================================================================
extern "C" void kernel_launch(void* const* d_in, const int* in_sizes, int n_in,
                              void* d_out, int out_size, void* d_ws, size_t ws_size,
                              hipStream_t stream)
{
    (void)in_sizes; (void)n_in; (void)out_size; (void)ws_size;
    // ---- inputs (setup_inputs dict order, nested dicts flattened depth-first)
    const float* x_in    = (const float*)d_in[0];   // (B,S,IN)
    const float* emb_w   = (const float*)d_in[1];   // (512,3000)
    const float* emb_b   = (const float*)d_in[2];
    const float* b0_ln1  = (const float*)d_in[3];
    const float* b0_up   = (const float*)d_in[4];   // (2048,512)
    const float* b0_cw   = (const float*)d_in[5];   // (1024,4)
    const float* b0_cb   = (const float*)d_in[6];
    const float* b0_qw   = (const float*)d_in[7];   // (256,4,4)
    const float* b0_kw   = (const float*)d_in[8];
    const float* b0_vw   = (const float*)d_in[9];
    const float* b0_igw  = (const float*)d_in[10];  // (4,3072)
    const float* b0_igb  = (const float*)d_in[11];
    const float* b0_fgw  = (const float*)d_in[12];
    const float* b0_fgb  = (const float*)d_in[13];
    const float* b0_nw   = (const float*)d_in[14];  // (1024)
    const float* b0_skip = (const float*)d_in[15];
    const float* b0_dw   = (const float*)d_in[16];  // (512,1024)
    const float* b1_ln1  = (const float*)d_in[17];
    const float* b1_cw   = (const float*)d_in[18];  // (512,4)
    const float* b1_cb   = (const float*)d_in[19];
    const float* b1_iw   = (const float*)d_in[20];  // (4,128,128)
    const float* b1_fw   = (const float*)d_in[21];
    const float* b1_zw   = (const float*)d_in[22];
    const float* b1_ow   = (const float*)d_in[23];
    const float* b1_rk   = (const float*)d_in[24];  // (4,128,512)
    const float* b1_rb   = (const float*)d_in[25];  // (4,4,128)
    const float* b1_nw   = (const float*)d_in[26];
    const float* b1_ln2  = (const float*)d_in[27];
    const float* b1_fu   = (const float*)d_in[28];  // (1408,512)
    const float* b1_fd   = (const float*)d_in[29];  // (512,704)
    const float* post_nw = (const float*)d_in[30];
    const float* fc_w    = (const float*)d_in[31];
    const float* fc_b    = (const float*)d_in[32];
    float* out = (float*)d_out;                     // (B,1) fp32

    const long long T = T_TOK;

    // ---- workspace bump allocator (256B aligned, deterministic)
    char* wsb = (char*)d_ws;
    size_t off = 0;
    auto alloc = [&](size_t bytes) -> void* {
        void* p = wsb + off;
        off = (off + bytes + 255) & ~(size_t)255;
        return p;
    };
    float* h    = (float*)alloc(T * EE * 4);       // residual stream
    float* lnA  = (float*)alloc(T * EE * 4);       // reused: ln0 -> ln2o -> post-ln
    float* xi   = (float*)alloc(T * 2 * INNER_ * 4); // [x_m | z]; reused: gi/gf/gz/go
    float* xc   = (float*)alloc(T * INNER_ * 4);   // mLSTM conv; reused: x1|x_c2
    float* q    = (float*)alloc(T * INNER_ * 4);   // reused: head-LN of attn out
    float* k    = (float*)alloc(T * INNER_ * 4);   // reused: h2
    float* v    = (float*)alloc(T * INNER_ * 4);   // reused: ys | hln
    float* hatt = (float*)alloc(T * INNER_ * 4);   // reused: geglu act
    float* u    = (float*)alloc(T * 2 * FF_ * 4);
    float* ig   = (float*)alloc((size_t)BB * NHM * SS * 4);
    float* fg   = (float*)alloc((size_t)BB * NHM * SS * 4);
    float* lfc  = (float*)alloc((size_t)BB * NHM * (SS + 1) * 4);
    bf16* pA  = (bf16*)alloc(T * IN_FP * 2);       // biggest A pack, reused
    bf16* pA2 = (bf16*)alloc(T * EE * 2);
    bf16* pA3 = (bf16*)alloc(T * EE * 2);
    bf16* pW  = (bf16*)alloc((size_t)EE * IN_FP * 2); // biggest W pack, reused

    auto pack = [&](bf16* dst, const float* src, int rows, int K, int Kp) {
        long long n = (long long)rows * Kp;
        pack_bf16_kernel<<<(unsigned)((n + 255) / 256), 256, 0, stream>>>(dst, src, rows, K, Kp);
    };
    auto gemm = [&](const bf16* A, int lda, const bf16* Wm, int ldb, float* C, int ldc,
                    const float* bias, const float* resid, int M, int N, int K) {
        dim3 g(N / 128, M / 128);
        gemm_bf16_wmma_kernel<<<g, 256, 0, stream>>>(A, lda, Wm, ldb, C, ldc, bias, resid, M, N, K);
    };
    auto ewgrid = [](long long n) { return (unsigned)((n + 255) / 256); };

    // ================= embedding: h = x @ emb_w^T + emb_b =================
    pack(pA, x_in, T_TOK, IN_F, IN_FP);
    pack(pW, emb_w, EE, IN_F, IN_FP);
    gemm(pA, IN_FP, pW, IN_FP, h, EE, emb_b, nullptr, T_TOK, EE, IN_FP);

    // ================= mLSTM block =================
    ln_kernel<<<T_TOK, 256, 0, stream>>>(lnA, h, b0_ln1, EE, 0);
    pack(pA2, lnA, T_TOK, EE, EE);
    pack(pW, b0_up, 2 * INNER_, EE, EE);
    gemm(pA2, EE, pW, EE, xi, 2 * INNER_, nullptr, nullptr, T_TOK, 2 * INNER_, EE);

    // causal conv + SiLU on x_m (strided view of xi)
    conv_silu_kernel<<<ewgrid(T * INNER_), 256, 0, stream>>>(xc, INNER_, xi, 2 * INNER_, b0_cw, b0_cb, INNER_, 1);

    // q,k from x_c ; v from x_m
    headwise44_kernel<<<ewgrid(T * NQK_), 256, 0, stream>>>(q, INNER_, xc, INNER_, b0_qw);
    headwise44_kernel<<<ewgrid(T * NQK_), 256, 0, stream>>>(k, INNER_, xc, INNER_, b0_kw);
    headwise44_kernel<<<ewgrid(T * NQK_), 256, 0, stream>>>(v, INNER_, xi, 2 * INNER_, b0_vw);

    igfg_kernel<<<T_TOK, 256, 0, stream>>>(ig, fg, q, k, v, b0_igw, b0_igb, b0_fgw, b0_fgb);
    lfc_kernel<<<(BB * NHM + 63) / 64, 64, 0, stream>>>(lfc, fg);

    size_t attn_lds = (size_t)(2 * SS * DHM + DHM + SS + SS + 256 + SS + SS + 1) * 4;
    mlstm_attn_kernel<<<BB * NHM, 256, attn_lds, stream>>>(hatt, q, k, v, ig, lfc);

    float* hlnA = q;   // reuse
    ln_kernel<<<T_TOK * NHM, 256, 0, stream>>>(hlnA, hatt, b0_nw, DHM, NHM);   // per-head LN

    float* h2 = k;     // reuse
    ew_h2_kernel<<<ewgrid(T * INNER_), 256, 0, stream>>>(h2, hlnA, xc, xi, b0_skip);

    pack(pA, h2, T_TOK, INNER_, INNER_);
    pack(pW, b0_dw, EE, INNER_, INNER_);
    gemm(pA, INNER_, pW, INNER_, h, EE, nullptr, h, T_TOK, EE, INNER_);   // h += mlstm out

    // ================= sLSTM block =================
    float* x1  = xc;            // reuse: ln(h)
    float* xc2 = xc + T * EE;   // conv+silu of x1
    ln_kernel<<<T_TOK, 256, 0, stream>>>(x1, h, b1_ln1, EE, 0);
    conv_silu_kernel<<<ewgrid(T * EE), 256, 0, stream>>>(xc2, EE, x1, EE, b1_cw, b1_cb, EE, 1);

    // grouped gate GEMMs (4 heads x 128x128): i,f from x_c2 ; z,o from x1
    float* gi = xi;
    float* gf = xi + T * EE;
    float* gz = xi + 2 * T * EE;
    float* go = xi + 3 * T * EE;
    pack(pA2, xc2, T_TOK, EE, EE);
    pack(pA3, x1,  T_TOK, EE, EE);
    const float* gw[4] = { b1_iw, b1_fw, b1_zw, b1_ow };
    float*       gd[4] = { gi, gf, gz, go };
    const bf16*  ga[4] = { pA2, pA2, pA3, pA3 };
    for (int g4 = 0; g4 < 4; ++g4) {
        pack(pW, gw[g4], NHS * DHS_, DHS_, DHS_);
        for (int hh = 0; hh < NHS; ++hh)
            gemm(ga[g4] + hh * DHS_, EE, pW + (size_t)hh * DHS_ * DHS_, DHS_,
                 gd[g4] + hh * DHS_, EE, nullptr, nullptr, T_TOK, DHS_, DHS_);
    }

    float* ys  = v;            // reuse
    float* hln = v + T * EE;   // reuse
    size_t scan_lds = (size_t)(DHS_ * 4 * DHS_ + DHS_ + 4 * DHS_) * 4; // ~258KB (CDNA5: 320KB/WG)
    slstm_scan_kernel<<<BB * NHS, 256, scan_lds, stream>>>(ys, gi, gf, gz, go, b1_rk, b1_rb);

    ln_kernel<<<T_TOK * NHS, 256, 0, stream>>>(hln, ys, b1_nw, DHS_, NHS);    // per-head LN
    ew_add_kernel<<<ewgrid(T * EE), 256, 0, stream>>>(h, hln, T * EE);        // residual

    // FFN (GELU-gated)
    ln_kernel<<<T_TOK, 256, 0, stream>>>(lnA, h, b1_ln2, EE, 0);
    pack(pA2, lnA, T_TOK, EE, EE);
    pack(pW, b1_fu, 2 * FF_, EE, EE);
    gemm(pA2, EE, pW, EE, u, 2 * FF_, nullptr, nullptr, T_TOK, 2 * FF_, EE);

    float* act = hatt; // reuse
    ew_geglu_kernel<<<ewgrid(T * FF_), 256, 0, stream>>>(act, u);
    pack(pA, act, T_TOK, FF_, FF_);
    pack(pW, b1_fd, EE, FF_, FF_);
    gemm(pA, FF_, pW, FF_, h, EE, nullptr, h, T_TOK, EE, FF_);                // h += ffn out

    // ================= head =================
    ln_kernel<<<T_TOK, 256, 0, stream>>>(lnA, h, post_nw, EE, 0);
    fc_softplus_kernel<<<BB, 256, 0, stream>>>(out, lnA, fc_w, fc_b);
}
